// RN_67242007986640
// MI455X (gfx1250) — matrix operand order repaired and checked by reading, not compiled, not run on hardware
//
#include <hip/hip_runtime.h>
#include <hip/hip_bf16.h>

typedef __attribute__((ext_vector_type(16))) __bf16 v16bf;
typedef __attribute__((ext_vector_type(8)))  __bf16 bf16x8;
typedef __attribute__((ext_vector_type(8)))  float  v8f;

#define CDIV(a,b) (((a)+(b)-1)/(b))

// ---------------- conv (stride2, pad1, k3) + relu ----------------
__global__ void conv_relu(const float* __restrict__ in, const float* __restrict__ w,
                          const float* __restrict__ bias, float* __restrict__ out,
                          int Cin, int Hin, int Hout) {
  int idx = blockIdx.x * blockDim.x + threadIdx.x;
  int total = 64 * 24 * Hout * Hout;
  if (idx >= total) return;
  int wo = idx % Hout; int t = idx / Hout;
  int ho = t % Hout;   t /= Hout;
  int co = t % 24;     int b = t / 24;
  float acc = bias[co];
  for (int ci = 0; ci < Cin; ++ci) {
    const float* ip = in + (size_t)(b * Cin + ci) * Hin * Hin;
    const float* wp = w + (co * Cin + ci) * 9;
    for (int kh = 0; kh < 3; ++kh) {
      int hi = 2 * ho - 1 + kh;
      if (hi < 0 || hi >= Hin) continue;
      for (int kw = 0; kw < 3; ++kw) {
        int wi = 2 * wo - 1 + kw;
        if (wi < 0 || wi >= Hin) continue;
        acc += ip[hi * Hin + wi] * wp[kh * 3 + kw];
      }
    }
  }
  out[idx] = fmaxf(acc, 0.f);
}

// ---------------- batchnorm stats: mean + rsqrt(var+eps) per channel ----------------
__global__ void bn_stats(const float* __restrict__ x, float* __restrict__ stats, int HW) {
  int c = blockIdx.x;            // 24 channels
  int N = 64 * HW;
  float s = 0.f, s2 = 0.f;
  for (int i = threadIdx.x; i < N; i += blockDim.x) {
    int b = i / HW, hw = i % HW;
    float v = x[(size_t)(b * 24 + c) * HW + hw];
    s += v; s2 += v * v;
  }
  __shared__ float sh[256], sh2[256];
  sh[threadIdx.x] = s; sh2[threadIdx.x] = s2;
  __syncthreads();
  for (int st = 128; st > 0; st >>= 1) {
    if ((int)threadIdx.x < st) { sh[threadIdx.x] += sh[threadIdx.x + st]; sh2[threadIdx.x] += sh2[threadIdx.x + st]; }
    __syncthreads();
  }
  if (threadIdx.x == 0) {
    float m = sh[0] / (float)N;
    float v = sh2[0] / (float)N - m * m;
    stats[c * 2] = m;
    stats[c * 2 + 1] = rsqrtf(v + 1e-5f);
  }
}

__global__ void bn_apply(float* __restrict__ x, const float* __restrict__ stats,
                         const float* __restrict__ gg, const float* __restrict__ bb, int HW) {
  int idx = blockIdx.x * blockDim.x + threadIdx.x;
  int total = 64 * 24 * HW;
  if (idx >= total) return;
  int c = (idx / HW) % 24;
  x[idx] = (x[idx] - stats[c * 2]) * stats[c * 2 + 1] * gg[c] + bb[c];
}

// ---------------- object tensor o = [x4 channels, row, col] ----------------
__global__ void build_o(const float* __restrict__ x4, float* __restrict__ o) {
  int idx = blockIdx.x * blockDim.x + threadIdx.x;
  if (idx >= 64 * 64 * 26) return;
  int c = idx % 26; int t = idx / 26;
  int m = t & 63;   int b = t >> 6;
  float v;
  if (c < 24)       v = x4[(size_t)(b * 24 + c) * 64 + m];
  else if (c == 24) v = (float)(m >> 3);
  else              v = (float)(m & 7);
  o[idx] = v;
}

// ---------------- per-object projections aT=o@Wi, bT=o@Wj ----------------
__global__ void proj_ab(const float* __restrict__ o, const float* __restrict__ gw1,
                        float* __restrict__ aT, float* __restrict__ bT) {
  int idx = blockIdx.x * blockDim.x + threadIdx.x;
  if (idx >= 64 * 64 * 256) return;
  int n = idx & 255; int m = (idx >> 8) & 63; int b = idx >> 14;
  const float* ob = o + (size_t)(b * 64 + m) * 26;
  float sa = 0.f, sb = 0.f;
  for (int c = 0; c < 26; ++c) {
    float ov = ob[c];
    sa += ov * gw1[c * 256 + n];
    sb += ov * gw1[(26 + c) * 256 + n];
  }
  aT[idx] = sa; bT[idx] = sb;
}

__global__ void proj_q(const float* __restrict__ qst, const float* __restrict__ gw1,
                       const float* __restrict__ gb1, float* __restrict__ q) {
  int idx = blockIdx.x * blockDim.x + threadIdx.x;
  if (idx >= 64 * 256) return;
  int n = idx & 255; int b = idx >> 8;
  float s = gb1[n];
  for (int c = 0; c < 11; ++c) s += qst[b * 11 + c] * gw1[(52 + c) * 256 + n];
  q[idx] = s;
}

// ---------------- pack a 256x256 f32 weight into WMMA B-fragment bf16 order ----------------
// out[((kb*16+nb)*32+lane)*16+e] = w[k*256+n], k = kb*32 + (lane>=16?8:0) + (e<8 ? e : e+8), n = nb*16 + (lane&15)
__global__ void pack_w(const float* __restrict__ w, __bf16* __restrict__ out) {
  int idx = blockIdx.x * blockDim.x + threadIdx.x;
  if (idx >= 8 * 16 * 32 * 16) return;
  int e    =  idx        & 15;
  int lane = (idx >> 4)  & 31;
  int nb   = (idx >> 9)  & 15;
  int kb   =  idx >> 13;
  int k = kb * 32 + ((lane >> 4) ? 8 : 0) + (e < 8 ? e : e + 8);
  int n = nb * 16 + (lane & 15);
  out[idx] = (__bf16)w[k * 256 + n];
}

__global__ void zero_f32(float* __restrict__ p, int n) {
  int i = blockIdx.x * blockDim.x + threadIdx.x;
  if (i < n) p[i] = 0.f;
}

// ---------------- fused relation network: 3 x (256x256 GEMM + bias + relu), sum over pairs ----------------
__global__ __launch_bounds__(128) void relation_wmma(
    const float* __restrict__ aT, const float* __restrict__ bT, const float* __restrict__ q,
    const __bf16* __restrict__ pW,
    const float* __restrict__ gb2, const float* __restrict__ gb3, const float* __restrict__ gb4,
    float* __restrict__ g) {
  __shared__ __align__(16) __bf16 hbuf[4][16][256];  // per-wave 16-row activation tile (bf16)
  __shared__ float gacc[256];
  int tid  = threadIdx.x;
  int wave = tid >> 5;
  int lane = tid & 31;
  int b    = blockIdx.x >> 6;
  int tile = ((blockIdx.x & 63) << 2) + wave;   // 0..255 -> 16 pairs each

  for (int i2 = tid; i2 < 256; i2 += 128) gacc[i2] = 0.f;
  __syncthreads();

  const float* aB = aT + (size_t)b * 64 * 256;
  const float* bB = bT + (size_t)b * 64 * 256;
  const float* qB = q  + (size_t)b * 256;

  // layer 1: h1[r][n] = relu(aT[j_r][n] + bT[i_r][n] + q[n]) -> bf16 in LDS
  for (int idx = lane; idx < 16 * 256; idx += 32) {
    int r = idx >> 8, n = idx & 255;
    int p = tile * 16 + r;
    int i = p >> 6, j = p & 63;
    float v = aB[j * 256 + n] + bB[i * 256 + n] + qB[n];
    hbuf[wave][r][n] = (__bf16)fmaxf(v, 0.f);
  }

  int rowsel = lane >> 4;       // A layout: lanes>=16 carry K+8 halves; D rows +8
  int col    = lane & 15;       // A row M / B,D column N (within tile)

  for (int layer = 0; layer < 3; ++layer) {
    // load all 8 A fragments (16x32 bf16 each) from this wave's LDS tile
    v16bf afrag[8];
#pragma unroll
    for (int kb = 0; kb < 8; ++kb) {
      int k0 = kb * 32 + (rowsel ? 8 : 0);
      const __bf16* hrow = &hbuf[wave][col][0];
      bf16x8 lo = *(const bf16x8*)(hrow + k0);       // K = k0..k0+7
      bf16x8 hi = *(const bf16x8*)(hrow + k0 + 16);  // K = k0+16..k0+23
#pragma unroll
      for (int e = 0; e < 8; ++e) { afrag[kb][e] = lo[e]; afrag[kb][e + 8] = hi[e]; }
    }

    const __bf16* wL   = pW + (size_t)layer * 8 * 16 * 512;
    const float*  bias = (layer == 0) ? gb2 : (layer == 1) ? gb3 : gb4;

#pragma unroll 1
    for (int nb = 0; nb < 16; ++nb) {
      v8f acc = {};
#pragma unroll
      for (int kb = 0; kb < 8; ++kb) {
        v16bf bfrag = *(const v16bf*)(wL + (size_t)(kb * 16 + nb) * 512 + lane * 16);
        acc = __builtin_amdgcn_wmma_f32_16x16x32_bf16(false, afrag[kb], false, bfrag,
                                                      (short)0, acc, false, false);
      }
      float bv = bias[nb * 16 + col];
      if (layer < 2) {
#pragma unroll
        for (int rr = 0; rr < 8; ++rr) {
          float v = fmaxf(acc[rr] + bv, 0.f);
          hbuf[wave][rr + (rowsel ? 8 : 0)][nb * 16 + col] = (__bf16)v;  // D layout -> next A
        }
      } else {
        float s = 0.f;
#pragma unroll
        for (int rr = 0; rr < 8; ++rr) s += fmaxf(acc[rr] + bv, 0.f);
        atomicAdd(&gacc[nb * 16 + col], s);   // both lane halves cover all 16 rows
      }
    }
  }
  __syncthreads();
  for (int i2 = tid; i2 < 256; i2 += 128)
    atomicAdd(&g[b * 256 + i2], gacc[i2]);
}

// ---------------- small dense layers ----------------
__global__ void fc(const float* __restrict__ in, const float* __restrict__ w,
                   const float* __restrict__ bias, float* __restrict__ out,
                   int K, int N, int dorelu) {
  int idx = blockIdx.x * blockDim.x + threadIdx.x;
  if (idx >= 64 * N) return;
  int n = idx % N, b = idx / N;
  float s = bias[n];
  for (int k = 0; k < K; ++k) s += in[b * K + k] * w[k * N + n];
  out[idx] = dorelu ? fmaxf(s, 0.f) : s;
}

__global__ void softmax10(const float* __restrict__ logits, float* __restrict__ out) {
  int b = blockIdx.x * blockDim.x + threadIdx.x;
  if (b >= 64) return;
  const float* l = logits + b * 10;
  float m = l[0];
  for (int i = 1; i < 10; ++i) m = fmaxf(m, l[i]);
  float e[10], s = 0.f;
  for (int i = 0; i < 10; ++i) { e[i] = expf(l[i] - m); s += e[i]; }
  for (int i = 0; i < 10; ++i) out[b * 10 + i] = e[i] / s;
}

// ---------------- workspace layout (bytes) ----------------
// Region X1 (25.2 MB) is reused for relation buffers after conv2 consumed it.
#define OFF_X1    ((size_t)0)
#define OFF_X2    ((size_t)25165824)
#define OFF_X3    (OFF_X2 + 6291456)
#define OFF_X4    (OFF_X3 + 1572864)
#define OFF_STATS (OFF_X4 + 393216)
// reuse inside X1:
#define OFF_AT ((size_t)0)
#define OFF_BT ((size_t)4194304)
#define OFF_Q  ((size_t)8388608)
#define OFF_O  ((size_t)8454144)
#define OFF_PW ((size_t)8912896)
#define OFF_G  ((size_t)9306112)
#define OFF_F1 ((size_t)9371648)
#define OFF_F2 ((size_t)9437184)
#define OFF_LG ((size_t)9502720)

extern "C" void kernel_launch(void* const* d_in, const int* in_sizes, int n_in,
                              void* d_out, int out_size, void* d_ws, size_t ws_size,
                              hipStream_t stream) {
  const float* img = (const float*)d_in[0];
  const float* qst = (const float*)d_in[1];
  const float* cw[4] = {(const float*)d_in[2],  (const float*)d_in[6],  (const float*)d_in[10], (const float*)d_in[14]};
  const float* cb[4] = {(const float*)d_in[3],  (const float*)d_in[7],  (const float*)d_in[11], (const float*)d_in[15]};
  const float* bg[4] = {(const float*)d_in[4],  (const float*)d_in[8],  (const float*)d_in[12], (const float*)d_in[16]};
  const float* bb[4] = {(const float*)d_in[5],  (const float*)d_in[9],  (const float*)d_in[13], (const float*)d_in[17]};
  const float* gw1 = (const float*)d_in[18]; const float* gb1 = (const float*)d_in[19];
  const float* gw2 = (const float*)d_in[20]; const float* gb2 = (const float*)d_in[21];
  const float* gw3 = (const float*)d_in[22]; const float* gb3 = (const float*)d_in[23];
  const float* gw4 = (const float*)d_in[24]; const float* gb4 = (const float*)d_in[25];
  const float* fw1 = (const float*)d_in[26]; const float* fb1 = (const float*)d_in[27];
  const float* fw2 = (const float*)d_in[28]; const float* fb2 = (const float*)d_in[29];
  const float* fw3 = (const float*)d_in[30]; const float* fb3 = (const float*)d_in[31];

  char* ws = (char*)d_ws;
  float* x[4]   = {(float*)(ws + OFF_X1), (float*)(ws + OFF_X2), (float*)(ws + OFF_X3), (float*)(ws + OFF_X4)};
  float* stats  = (float*)(ws + OFF_STATS);
  float* aTb    = (float*)(ws + OFF_AT);
  float* bTb    = (float*)(ws + OFF_BT);
  float* qb     = (float*)(ws + OFF_Q);
  float* ob     = (float*)(ws + OFF_O);
  __bf16* pW    = (__bf16*)(ws + OFF_PW);
  float* gbuf   = (float*)(ws + OFF_G);
  float* f1     = (float*)(ws + OFF_F1);
  float* f2     = (float*)(ws + OFF_F2);
  float* logits = (float*)(ws + OFF_LG);
  float* probs  = (float*)d_out;

  // conv trunk: 128 -> 64 -> 32 -> 16 -> 8
  const int Cin[4]  = {3, 24, 24, 24};
  const int Hin[4]  = {128, 64, 32, 16};
  const int Hout[4] = {64, 32, 16, 8};
  const float* src = img;
  for (int L = 0; L < 4; ++L) {
    int total = 64 * 24 * Hout[L] * Hout[L];
    conv_relu<<<CDIV(total, 256), 256, 0, stream>>>(src, cw[L], cb[L], x[L], Cin[L], Hin[L], Hout[L]);
    bn_stats<<<24, 256, 0, stream>>>(x[L], stats, Hout[L] * Hout[L]);
    bn_apply<<<CDIV(total, 256), 256, 0, stream>>>(x[L], stats, bg[L], bb[L], Hout[L] * Hout[L]);
    src = x[L];
  }

  // objects + projections
  build_o<<<CDIV(64 * 64 * 26, 256), 256, 0, stream>>>(x[3], ob);
  proj_ab<<<CDIV(64 * 64 * 256, 256), 256, 0, stream>>>(ob, gw1, aTb, bTb);
  proj_q<<<CDIV(64 * 256, 256), 256, 0, stream>>>(qst, gw1, gb1, qb);

  // pack 3 weight matrices into WMMA B-fragment bf16 order
  pack_w<<<CDIV(65536, 256), 256, 0, stream>>>(gw2, pW);
  pack_w<<<CDIV(65536, 256), 256, 0, stream>>>(gw3, pW + 65536);
  pack_w<<<CDIV(65536, 256), 256, 0, stream>>>(gw4, pW + 131072);

  // fused relation network (all WMMA)
  zero_f32<<<CDIV(64 * 256, 256), 256, 0, stream>>>(gbuf, 64 * 256);
  relation_wmma<<<4096, 128, 0, stream>>>(aTb, bTb, qb, pW, gb2, gb3, gb4, gbuf);

  // f-MLP + softmax
  fc<<<CDIV(64 * 256, 256), 256, 0, stream>>>(gbuf, fw1, fb1, f1, 256, 256, 1);
  fc<<<CDIV(64 * 256, 256), 256, 0, stream>>>(f1, fw2, fb2, f2, 256, 256, 1);
  fc<<<CDIV(64 * 10, 256), 256, 0, stream>>>(f2, fw3, fb3, logits, 256, 10, 0);
  softmax10<<<1, 64, 0, stream>>>(logits, probs);
}
